// _RPN_49323404427574
// MI455X (gfx1250) — matrix operand order repaired and checked by reading, not compiled, not run on hardware
//
#include <hip/hip_runtime.h>

typedef __bf16 bf16;
typedef __attribute__((ext_vector_type(16))) __bf16 v16bf;
typedef __attribute__((ext_vector_type(8)))  float  v8f;

#define B_    8
#define H_    64
#define W_    100
#define HP    66              // padded
#define WP    102             // padded
#define C_    512
#define NSP   (H_*W_)          // 6400 spatial / batch
#define KTOT  (C_*9)           // 4608
#define NKT   (KTOT/32)        // 144 k-tiles for conv
#define NMT   (C_/16)          // 32 m-tiles (16 rows) for conv weights
#define NANCH 9
#define NPROP (NSP*NANCH)      // 57600 per batch
#define PRE_NMS 6000
#define POST_NMS 300
#define SORTN 8192
#define NTOT  (B_*NSP)         // 51200 spatial total

// ---------------- weight repack: conv 3x3 -> WMMA A fragments ----------------
// A-fragment packing used here (contiguous k, not AMD's table order): lane<16: row=lane, K in {0..15};
// lane>=16: row=lane-16, K in {16..31}.  K ordering: K = g*512 + ci, g=ky*3+kx.
__device__ __forceinline__ int koff_for(int lane, int j) {
  if (lane < 16) return j;
  return 16 + j;
}

__global__ void repack_conv_w(const float* __restrict__ w, bf16* __restrict__ apack) {
  int t = blockIdx.x * 256 + threadIdx.x;           // (mt,kt,lane)
  if (t >= NMT * NKT * 32) return;
  int lane = t & 31;
  int kt   = (t >> 5) % NKT;
  int mt   = t / (NKT * 32);
  int row  = mt * 16 + (lane & 15);
  bf16* dst = apack + (size_t)t * 16;
  for (int j = 0; j < 16; ++j) {
    int K  = kt * 32 + koff_for(lane, j);
    int g  = K >> 9;            // K/512
    int ci = K & 511;
    int ky = g / 3, kx = g % 3;
    dst[j] = (bf16)w[(((size_t)row * C_ + ci) * 3 + ky) * 3 + kx];
  }
}

// combined 1x1 head weights: rows 0..17 = cls_w, 18..53 = bbox_w, 54..63 = 0. K = ci.
__global__ void repack_head_w(const float* __restrict__ cw, const float* __restrict__ bw,
                              bf16* __restrict__ apack2) {
  int t = blockIdx.x * 256 + threadIdx.x;           // 4 mt * 16 kt * 32 lanes
  if (t >= 4 * 16 * 32) return;
  int lane = t & 31;
  int kt   = (t >> 5) & 15;
  int mt   = t >> 9;
  int row  = mt * 16 + (lane & 15);
  bf16* dst = apack2 + (size_t)t * 16;
  for (int j = 0; j < 16; ++j) {
    int ci = kt * 32 + koff_for(lane, j);
    float v = 0.f;
    if (row < 18)      v = cw[(size_t)row * C_ + ci];
    else if (row < 54) v = bw[(size_t)(row - 18) * C_ + ci];
    dst[j] = (bf16)v;
  }
}

// ---------------- NCHW f32 -> padded NHWC bf16 (LDS-tiled transpose) ----------------
// grid: (NTOT/64, C_/64); block 256.  Border of xp must be pre-zeroed (memset).
__global__ __launch_bounds__(256) void nchw_to_nhwc_pad(const float* __restrict__ x,
                                                        bf16* __restrict__ xp) {
  __shared__ float tile[64][65];
  int n0  = blockIdx.x * 64;          // 64 | NSP, so tile never straddles batches
  int ci0 = blockIdx.y * 64;
  int tx = threadIdx.x & 63;
  int ty = threadIdx.x >> 6;          // 0..3
  int b  = n0 / NSP;
  int nl0 = n0 - b * NSP;
#pragma unroll 4
  for (int k = 0; k < 16; ++k) {
    int cil = ty + 4 * k;
    tile[cil][tx] = x[((size_t)b * C_ + ci0 + cil) * NSP + (nl0 + tx)];
  }
  __syncthreads();
#pragma unroll 4
  for (int k = 0; k < 16; ++k) {
    int nl = nl0 + ty + 4 * k;
    int y = nl / W_, xc = nl - y * W_;
    bf16* dst = xp + (((size_t)b * HP + y + 1) * WP + (xc + 1)) * C_ + ci0;
    dst[tx] = (bf16)tile[tx][ty + 4 * k];
  }
}

// ---------------- 3x3 conv + bias + ReLU via WMMA (padded input, no branches) --------
// wave = b*800 + nt*8 + mt ; 64(M=Cout) x 64(N=spatial) per wave
__global__ __launch_bounds__(256) void conv3x3_wmma(const bf16* __restrict__ X,
                                                    const bf16* __restrict__ apack,
                                                    const float* __restrict__ bias,
                                                    bf16* __restrict__ Y) {
  __shared__ bf16 ctile[8][64][64];
  int warp = threadIdx.x >> 5;
  int wave = (blockIdx.x * 256 + threadIdx.x) >> 5;
  int lane = threadIdx.x & 31;
  int mt = wave & 7;
  int nt = (wave >> 3) % 100;
  int b  = wave / 800;
  int n0 = nt * 64;
  int colh = lane & 15;
  bool hi  = lane >= 16;

  // loop-invariant per-fragment pixel base pointers (interior of padded buffer)
  const bf16* pj[4];
#pragma unroll
  for (int j = 0; j < 4; ++j) {
    int n = n0 + j * 16 + colh;
    int y = n / W_, x = n - y * W_;
    pj[j] = X + (((size_t)b * HP + y + 1) * WP + (x + 1)) * C_ + (hi ? 16 : 0);
  }
  const bf16* ap = apack + ((size_t)(mt * 4) * NKT * 32 + lane) * 16;

  int goffs[9];
#pragma unroll
  for (int g = 0; g < 9; ++g) goffs[g] = ((g / 3 - 1) * WP + (g % 3 - 1)) * C_;

  v8f acc[4][4];
  v8f zf = {};
#pragma unroll
  for (int i = 0; i < 4; ++i)
#pragma unroll
    for (int j = 0; j < 4; ++j) acc[i][j] = zf;

  for (int g = 0; g < 9; ++g) {
    const bf16* bp0 = pj[0] + goffs[g];
    const bf16* bp1 = pj[1] + goffs[g];
    const bf16* bp2 = pj[2] + goffs[g];
    const bf16* bp3 = pj[3] + goffs[g];
    const bf16* ag  = ap + (size_t)g * 16 * 512;
    if (g < 8) {   // pull next halo group toward the WGP while this one computes
      __builtin_prefetch(pj[0] + goffs[g + 1], 0, 1);
      __builtin_prefetch(pj[1] + goffs[g + 1], 0, 1);
      __builtin_prefetch(pj[2] + goffs[g + 1], 0, 1);
      __builtin_prefetch(pj[3] + goffs[g + 1], 0, 1);
    }
#pragma unroll 4
    for (int t = 0; t < 16; ++t) {
      v16bf Bf0 = *(const v16bf*)(bp0 + t * 32);
      v16bf Bf1 = *(const v16bf*)(bp1 + t * 32);
      v16bf Bf2 = *(const v16bf*)(bp2 + t * 32);
      v16bf Bf3 = *(const v16bf*)(bp3 + t * 32);
      v16bf Af[4];
#pragma unroll
      for (int i = 0; i < 4; ++i)
        Af[i] = *(const v16bf*)(ag + (size_t)i * NKT * 512 + t * 512);
#pragma unroll
      for (int i = 0; i < 4; ++i) {
        acc[i][0] = __builtin_amdgcn_wmma_f32_16x16x32_bf16(false, Af[i], false, Bf0, (short)0, acc[i][0], false, false);
        acc[i][1] = __builtin_amdgcn_wmma_f32_16x16x32_bf16(false, Af[i], false, Bf1, (short)0, acc[i][1], false, false);
        acc[i][2] = __builtin_amdgcn_wmma_f32_16x16x32_bf16(false, Af[i], false, Bf2, (short)0, acc[i][2], false, false);
        acc[i][3] = __builtin_amdgcn_wmma_f32_16x16x32_bf16(false, Af[i], false, Bf3, (short)0, acc[i][3], false, false);
      }
    }
  }

  // epilogue: pack 8 consecutive-co bf16 per fragment -> one ds_store_b128 each,
  // then stream the 64x64 tile out as coalesced global_store_b128 rows.
  int rowh = hi ? 8 : 0;
#pragma unroll
  for (int i = 0; i < 4; ++i) {
#pragma unroll
    for (int j = 0; j < 4; ++j) {
      union { bf16 h[8]; uint4 u; } pk;
#pragma unroll
      for (int v = 0; v < 8; ++v) {
        int co = mt * 64 + i * 16 + v + rowh;
        float val = acc[i][j][v] + bias[co];
        pk.h[v] = (bf16)(val > 0.f ? val : 0.f);
      }
      *(uint4*)&ctile[warp][j * 16 + colh][i * 16 + rowh] = pk.u;
    }
  }
  int rsub = lane >> 3;            // 0..3 : row within 4-row group
  int csub = (lane & 7) * 8;       // 8 bf16 = 16B chunk within row
#pragma unroll
  for (int s = 0; s < 16; ++s) {
    int row = s * 4 + rsub;
    uint4 vdat = *(const uint4*)&ctile[warp][row][csub];
    *(uint4*)(Y + ((size_t)b * NSP + n0 + row) * C_ + mt * 64 + csub) = vdat;
  }
}

// ---------------- 1x1 head conv (54 out channels padded to 64) ----------------
__global__ __launch_bounds__(256) void head1x1_wmma(const bf16* __restrict__ C1,
                                                    const bf16* __restrict__ apack2,
                                                    const float* __restrict__ cls_b,
                                                    const float* __restrict__ bbox_b,
                                                    float* __restrict__ hd) {
  int wave = (blockIdx.x * 256 + threadIdx.x) >> 5;   // 0..799
  int lane = threadIdx.x & 31;
  int n0 = wave * 64;
  int colh = lane & 15;
  bool hi  = lane >= 16;

  v8f acc[4][4];
  v8f zf = {};
#pragma unroll
  for (int i = 0; i < 4; ++i)
#pragma unroll
    for (int j = 0; j < 4; ++j) acc[i][j] = zf;

  const bf16* bbase = C1 + (size_t)n0 * C_ + (hi ? 16 : 0);
  const bf16* abase = apack2 + (size_t)lane * 16;
  for (int kt = 0; kt < 16; ++kt) {
    v16bf Bf[4];
#pragma unroll
    for (int j = 0; j < 4; ++j)
      Bf[j] = *(const v16bf*)(bbase + (size_t)(j * 16 + colh) * C_ + kt * 32);
    v16bf Af[4];
#pragma unroll
    for (int i = 0; i < 4; ++i)
      Af[i] = *(const v16bf*)(abase + (size_t)(i * 16 + kt) * 512);
#pragma unroll
    for (int i = 0; i < 4; ++i)
#pragma unroll
      for (int j = 0; j < 4; ++j)
        acc[i][j] = __builtin_amdgcn_wmma_f32_16x16x32_bf16(
            false, Af[i], false, Bf[j], (short)0, acc[i][j], false, false);
  }

  int rowh = hi ? 8 : 0;
#pragma unroll
  for (int i = 0; i < 4; ++i) {
#pragma unroll
    for (int v = 0; v < 8; ++v) {
      int ch = i * 16 + v + rowh;
      if (ch >= 54) continue;
      float bs = (ch < 18) ? cls_b[ch] : bbox_b[ch - 18];
#pragma unroll
      for (int j = 0; j < 4; ++j) {
        int n = n0 + j * 16 + colh;
        hd[(size_t)n * 64 + ch] = acc[i][j][v] + bs;
      }
    }
  }
}

// ---------------- softmax over channel pairs -> prob output + scores ----------------
__global__ void softmax_kern(const float* __restrict__ hd, float* __restrict__ prob,
                             float* __restrict__ scores) {
  int t = blockIdx.x * 256 + threadIdx.x;           // b*6400*9
  if (t >= B_ * NSP * NANCH) return;
  int a = t % NANCH;
  int n = (t / NANCH) % NSP;
  int b = t / (NANCH * NSP);
  const float* h = hd + ((size_t)b * NSP + n) * 64;
  float c0 = h[a], c1 = h[9 + a];
  float m = fmaxf(c0, c1);
  float e0 = __expf(c0 - m), e1 = __expf(c1 - m);
  float inv = 1.f / (e0 + e1);
  prob[((size_t)b * 18 + a) * NSP + n]     = e0 * inv;
  prob[((size_t)b * 18 + 9 + a) * NSP + n] = e1 * inv;
  scores[(size_t)b * NPROP + n * NANCH + a] = e1 * inv;
}

// ---------------- anchor decode + clip ----------------
__global__ void box_kern(const float* __restrict__ hd, const float* __restrict__ im_info,
                         float* __restrict__ boxes) {
  int t = blockIdx.x * 256 + threadIdx.x;
  if (t >= B_ * NPROP) return;
  int i = t % NPROP, b = t / NPROP;
  int a = i % NANCH, n = i / NANCH;
  int h = n / W_, w = n - h * W_;
  int r = a / 3, s = a % 3;
  float ratio = (r == 0) ? 0.5f : (r == 1) ? 1.0f : 2.0f;
  float scale = (s == 0) ? 8.0f : (s == 1) ? 16.0f : 32.0f;
  float wsa = rintf(sqrtf(256.0f / ratio));
  float hsa = rintf(wsa * ratio);
  float ww = wsa * scale, hh = hsa * scale;
  float sx = w * 16.0f, sy = h * 16.0f, ctr = 7.5f;
  float ax1 = sx + ctr - 0.5f * (ww - 1.f);
  float ay1 = sy + ctr - 0.5f * (hh - 1.f);
  float ax2 = sx + ctr + 0.5f * (ww - 1.f);
  float ay2 = sy + ctr + 0.5f * (hh - 1.f);
  float aw = ax2 - ax1 + 1.f, ah = ay2 - ay1 + 1.f;
  float acx = ax1 + 0.5f * aw, acy = ay1 + 0.5f * ah;
  const float* d = hd + ((size_t)b * NSP + n) * 64 + 18 + a * 4;
  float pcx = d[0] * aw + acx, pcy = d[1] * ah + acy;
  float pw = __expf(d[2]) * aw, ph = __expf(d[3]) * ah;
  float imH = im_info[b * 3 + 0] - 1.f, imW = im_info[b * 3 + 1] - 1.f;
  float x1 = fminf(fmaxf(pcx - 0.5f * pw, 0.f), imW);
  float y1 = fminf(fmaxf(pcy - 0.5f * ph, 0.f), imH);
  float x2 = fminf(fmaxf(pcx + 0.5f * pw, 0.f), imW);
  float y2 = fminf(fmaxf(pcy + 0.5f * ph, 0.f), imH);
  float* o = boxes + (size_t)t * 4;
  o[0] = x1; o[1] = y1; o[2] = x2; o[3] = y2;
}

// ---------------- per-batch top-6000 select + bitonic sort (descending) ----------------
__global__ __launch_bounds__(1024) void topk_sort(const float* __restrict__ scores,
                                                  const float* __restrict__ boxes,
                                                  float* __restrict__ top_boxes,
                                                  float* __restrict__ top_scores) {
  __shared__ unsigned int skey[SORTN];
  __shared__ unsigned int sidx[SORTN];
  unsigned int* hist = skey;              // 2048 bins, dead before skey is written
  unsigned int* cnts = sidx + SORTN - 4;  // overwritten by pad step after use
  int b = blockIdx.x, tid = threadIdx.x;
  const float* sc = scores + (size_t)b * NPROP;

  for (int i = tid; i < 2048; i += 1024) hist[i] = 0;
  __syncthreads();
  for (int i = tid; i < NPROP; i += 1024)
    atomicAdd(&hist[__float_as_uint(sc[i]) >> 21], 1u);
  __syncthreads();
  if (tid == 0) {
    unsigned cum = 0; int t = 2047;
    for (; t > 0; --t) { if (cum + hist[t] >= PRE_NMS) break; cum += hist[t]; }
    cnts[0] = cum; cnts[1] = (unsigned)t; cnts[2] = 0; cnts[3] = 0;
  }
  __syncthreads();
  unsigned chi = cnts[0]; int tb = (int)cnts[1];
  unsigned need = PRE_NMS - chi;
  __syncthreads();
  for (int i = tid; i < NPROP; i += 1024) {
    unsigned k = __float_as_uint(sc[i]);
    int bin = (int)(k >> 21);
    if (bin > tb) {
      unsigned p = atomicAdd(&cnts[2], 1u);
      skey[p] = k; sidx[p] = (unsigned)i;
    } else if (bin == tb) {
      unsigned q = atomicAdd(&cnts[3], 1u);
      if (q < need) { skey[chi + q] = k; sidx[chi + q] = (unsigned)i; }
    }
  }
  __syncthreads();
  for (int i = PRE_NMS + tid; i < SORTN; i += 1024) { skey[i] = 0u; sidx[i] = 0u; }
  __syncthreads();
  // bitonic sort, descending by key
  for (int k = 2; k <= SORTN; k <<= 1) {
    for (int j = k >> 1; j > 0; j >>= 1) {
      for (int i = tid; i < SORTN; i += 1024) {
        int l = i ^ j;
        if (l > i) {
          bool dir = ((i & k) == 0);
          unsigned a = skey[i], c = skey[l];
          if ((a < c) == dir) {
            skey[i] = c; skey[l] = a;
            unsigned ti = sidx[i]; sidx[i] = sidx[l]; sidx[l] = ti;
          }
        }
      }
      __syncthreads();
    }
  }
  for (int r = tid; r < PRE_NMS; r += 1024) {
    unsigned idx = sidx[r];
    const float* bx = boxes + ((size_t)b * NPROP + idx) * 4;
    float* ob = top_boxes + ((size_t)b * PRE_NMS + r) * 4;
    ob[0] = bx[0]; ob[1] = bx[1]; ob[2] = bx[2]; ob[3] = bx[3];
    top_scores[(size_t)b * PRE_NMS + r] = __uint_as_float(skey[r]);
  }
}

// ---------------- per-batch greedy NMS + compaction to 300 ROIs ----------------
__global__ __launch_bounds__(1024) void nms_select(const float* __restrict__ top_boxes,
                                                   const float* __restrict__ top_scores,
                                                   float* __restrict__ rois,
                                                   float* __restrict__ roi_scores) {
  __shared__ unsigned char kept[PRE_NMS];
  __shared__ int flag;
  int b = blockIdx.x, tid = threadIdx.x;
  const float* tbx = top_boxes + (size_t)b * PRE_NMS * 4;
  for (int i = tid; i < POST_NMS * 5; i += 1024) rois[(size_t)b * POST_NMS * 5 + i] = 0.f;
  for (int i = tid; i < POST_NMS; i += 1024)     roi_scores[(size_t)b * POST_NMS + i] = 0.f;
  if (tid == 0) flag = 0;
  __syncthreads();
  for (int i = 0; i < PRE_NMS; ++i) {
    float ix1 = tbx[i * 4], iy1 = tbx[i * 4 + 1], ix2 = tbx[i * 4 + 2], iy2 = tbx[i * 4 + 3];
    float iarea = (ix2 - ix1 + 1.f) * (iy2 - iy1 + 1.f);
    for (int j = tid; j < i; j += 1024) {
      if (kept[j]) {
        float jx1 = tbx[j * 4], jy1 = tbx[j * 4 + 1], jx2 = tbx[j * 4 + 2], jy2 = tbx[j * 4 + 3];
        float xx1 = fmaxf(ix1, jx1), yy1 = fmaxf(iy1, jy1);
        float xx2 = fminf(ix2, jx2), yy2 = fminf(iy2, jy2);
        float w = fmaxf(0.f, xx2 - xx1 + 1.f), h = fmaxf(0.f, yy2 - yy1 + 1.f);
        float inter = w * h;
        float jarea = (jx2 - jx1 + 1.f) * (jy2 - jy1 + 1.f);
        if (inter / (iarea + jarea - inter) > 0.7f) flag = 1;
      }
    }
    __syncthreads();
    if (tid == 0) { kept[i] = (flag == 0) ? 1 : 0; flag = 0; }
    __syncthreads();
  }
  if (tid == 0) {
    int rank = 0;
    for (int i = 0; i < PRE_NMS && rank < POST_NMS; ++i) {
      if (kept[i]) {
        float* r = rois + ((size_t)b * POST_NMS + rank) * 5;
        r[0] = (float)b;
        r[1] = tbx[i * 4]; r[2] = tbx[i * 4 + 1]; r[3] = tbx[i * 4 + 2]; r[4] = tbx[i * 4 + 3];
        roi_scores[(size_t)b * POST_NMS + rank] = top_scores[(size_t)b * PRE_NMS + i];
        ++rank;
      }
    }
  }
}

extern "C" void kernel_launch(void* const* d_in, const int* in_sizes, int n_in,
                              void* d_out, int out_size, void* d_ws, size_t ws_size,
                              hipStream_t stream) {
  (void)in_sizes; (void)n_in; (void)out_size; (void)ws_size;
  const float* base_feat = (const float*)d_in[0];
  const float* im_info   = (const float*)d_in[1];
  const float* conv_w    = (const float*)d_in[2];
  const float* conv_b    = (const float*)d_in[3];
  const float* cls_w     = (const float*)d_in[4];
  const float* cls_b     = (const float*)d_in[5];
  const float* bbox_w    = (const float*)d_in[6];
  const float* bbox_b    = (const float*)d_in[7];

  char* ws = (char*)d_ws;
  size_t off = 0;
  auto alloc = [&](size_t bytes) { size_t cur = off; off = (off + bytes + 255) & ~(size_t)255; return cur; };
  const size_t xpad_bytes = (size_t)B_ * HP * WP * C_ * 2;
  bf16*  apack   = (bf16*)(ws + alloc((size_t)NMT * NKT * 32 * 16 * 2));
  bf16*  apack2  = (bf16*)(ws + alloc((size_t)4 * 16 * 32 * 16 * 2));
  bf16*  Xpad    = (bf16*)(ws + alloc(xpad_bytes));
  bf16*  conv1   = (bf16*)(ws + alloc((size_t)NTOT * C_ * 2));
  float* hd      = (float*)(ws + alloc((size_t)NTOT * 64 * 4));
  float* scores  = (float*)(ws + alloc((size_t)B_ * NPROP * 4));
  float* boxes   = (float*)(ws + alloc((size_t)B_ * NPROP * 4 * 4));
  float* top_bx  = (float*)(ws + alloc((size_t)B_ * PRE_NMS * 4 * 4));
  float* top_sc  = (float*)(ws + alloc((size_t)B_ * PRE_NMS * 4));

  float* out        = (float*)d_out;
  float* rois       = out;                       // 8*300*5
  float* roi_scores = out + B_ * POST_NMS * 5;   // 8*300
  float* prob       = out + B_ * POST_NMS * 6;   // 8*18*64*100

  hipMemsetAsync(Xpad, 0, xpad_bytes, stream);   // zero halo (interior rewritten below)
  repack_conv_w<<<(NMT * NKT * 32 + 255) / 256, 256, 0, stream>>>(conv_w, apack);
  repack_head_w<<<(4 * 16 * 32 + 255) / 256, 256, 0, stream>>>(cls_w, bbox_w, apack2);
  {
    dim3 g(NTOT / 64, C_ / 64);
    nchw_to_nhwc_pad<<<g, 256, 0, stream>>>(base_feat, Xpad);
  }
  conv3x3_wmma<<<800, 256, 0, stream>>>(Xpad, apack, conv_b, conv1);
  head1x1_wmma<<<100, 256, 0, stream>>>(conv1, apack2, cls_b, bbox_b, hd);
  softmax_kern<<<(B_ * NSP * NANCH + 255) / 256, 256, 0, stream>>>(hd, prob, scores);
  box_kern<<<(B_ * NPROP + 255) / 256, 256, 0, stream>>>(hd, im_info, boxes);
  topk_sort<<<B_, 1024, 0, stream>>>(scores, boxes, top_bx, top_sc);
  nms_select<<<B_, 1024, 0, stream>>>(top_bx, top_sc, rois, roi_scores);
}
